// MoE_84868553769174
// MI455X (gfx1250) — matrix operand order repaired and checked by reading, not compiled, not run on hardware
//
#include <hip/hip_runtime.h>
#include <hip/hip_bf16.h>
#include <math.h>

typedef __attribute__((ext_vector_type(16))) __bf16 v16bf;
typedef __attribute__((ext_vector_type(8)))  float  v8f;

#define NTOK    4096
#define DIM     1024
#define HALF_IN 2048     // gate width == inp width
#define NOUTK   2048     // W_out K dim
#define NEXP    8
#define KC      64       // K elements staged in LDS per step
#define LDK     72       // padded LDS row stride (elements) -> 144B, 16B aligned

// workspace layout (bytes)
#define WS_TOKS  1024
#define WS_WL    (1024 + NEXP*4096*4)
#define WS_ACT   (512*1024)

union FragU { uint4 q[2]; v16bf v; };

static __device__ __forceinline__ unsigned short f2bf(float f) {
  return __builtin_bit_cast(unsigned short, (__bf16)f);
}

// 32-bit LDS byte address of a __shared__ object (for async-to-LDS VDST)
static __device__ __forceinline__ unsigned lds_addr32(const void* p) {
  return (unsigned)(size_t)(__attribute__((address_space(3))) const void*)p;
}

__global__ __launch_bounds__(256) void k_init(float* __restrict__ y, int* __restrict__ counts) {
  int idx = blockIdx.x * 256 + threadIdx.x;
  ((float4*)y)[idx] = make_float4(0.f, 0.f, 0.f, 0.f);   // 4096*256*4 = 4M floats
  if (blockIdx.x == 0 && threadIdx.x < NEXP) counts[threadIdx.x] = 0;
}

__global__ __launch_bounds__(256) void k_router(const float* __restrict__ x,
                                                const float* __restrict__ Wr,
                                                int* __restrict__ counts,
                                                int* __restrict__ toks,
                                                float* __restrict__ wl) {
  __shared__ float wr_s[NEXP * DIM];                      // 32 KB
  int tid = threadIdx.x;
  for (int i = tid; i < NEXP * DIM; i += 256) wr_s[i] = Wr[i];
  __syncthreads();

  int wave = tid >> 5, lane = tid & 31;
  int t = blockIdx.x * 8 + wave;

  float xv[32];
  const float* xp = x + (size_t)t * DIM;
#pragma unroll
  for (int i = 0; i < 32; i++) xv[i] = xp[i * 32 + lane];

  float lg[NEXP];
#pragma unroll
  for (int e = 0; e < NEXP; e++) {
    float acc = 0.f;
#pragma unroll
    for (int i = 0; i < 32; i++) acc += xv[i] * wr_s[e * DIM + i * 32 + lane];
#pragma unroll
    for (int off = 16; off > 0; off >>= 1) acc += __shfl_xor(acc, off, 32);
    lg[e] = acc;
  }
  // stable top-2 (ties -> lowest index, matches stable argsort of -logits)
  int e0 = 0; float l0 = lg[0];
#pragma unroll
  for (int e = 1; e < NEXP; e++) if (lg[e] > l0) { l0 = lg[e]; e0 = e; }
  int e1 = -1; float l1 = -3.4e38f;
#pragma unroll
  for (int e = 0; e < NEXP; e++) if (e != e0 && lg[e] > l1) { l1 = lg[e]; e1 = e; }
  float b  = __expf(l1 - l0);
  float w0 = 1.f / (1.f + b);
  float w1 = b   / (1.f + b);

  if (lane == 0) {
    int p0 = atomicAdd(&counts[e0], 1);
    toks[e0 * 4096 + p0] = t;  wl[e0 * 4096 + p0] = w0;
    int p1 = atomicAdd(&counts[e1], 1);
    toks[e1 * 4096 + p1] = t;  wl[e1 * 4096 + p1] = w1;
  }
}

// h = silu(x@Win_gate) * (x@Win_inp), per-expert compacted, bf16 out
__global__ __launch_bounds__(256) void k_ffn1(const float* __restrict__ x,
                                              const float* __restrict__ Win,
                                              const int* __restrict__ counts,
                                              const int* __restrict__ toks,
                                              unsigned short* __restrict__ act) {
  int e = blockIdx.z, jn = blockIdx.x * 128, rt = blockIdx.y;
  int cnt = counts[e];
  if (rt * 128 >= cnt) return;
  int base = 0;
  for (int q = 0; q < e; q++) base += (counts[q] + 127) & ~127;
  int slot0 = base + rt * 128;

  __shared__ unsigned short xs[128 * LDK];
  __shared__ unsigned short gs[128 * LDK];
  __shared__ unsigned short is_[128 * LDK];
  __shared__ int ts[128];

  int tid = threadIdx.x, wave = tid >> 5, lane = tid & 31;
  if (tid < 128) {
    int idx = rt * 128 + tid;
    ts[tid] = (idx < cnt) ? toks[e * 4096 + idx] : 0;
  }
  int m0 = (wave >> 1) * 32, n0 = (wave & 1) * 64;   // 8 wave patches of 32x64

  v8f cg[2][4], ci[2][4];
#pragma unroll
  for (int mi = 0; mi < 2; mi++)
#pragma unroll
    for (int ni = 0; ni < 4; ni++) { cg[mi][ni] = (v8f)0.0f; ci[mi][ni] = (v8f)0.0f; }

  const float* winG = Win + ((size_t)e << 22) + jn;       // e*1024*4096
  const float* winI = winG + HALF_IN;

  for (int k0 = 0; k0 < DIM; k0 += KC) {
    __syncthreads();
    // stage x tile: 128 rows x 64 k, f32 -> bf16
#pragma unroll
    for (int it = 0; it < 8; it++) {
      int idx = it * 1024 + tid * 4;
      int row = idx >> 6, k = idx & 63;
      float4 v = *(const float4*)(x + (size_t)ts[row] * DIM + k0 + k);
      unsigned short* dst = &xs[row * LDK + k];
      dst[0] = f2bf(v.x); dst[1] = f2bf(v.y); dst[2] = f2bf(v.z); dst[3] = f2bf(v.w);
    }
    // stage Win tiles transposed to [n][k]
#pragma unroll
    for (int it = 0; it < 16; it++) {
      int kl = it * 4 + (tid >> 6);
      int nl = (tid & 63) * 2;
      float2 g  = *(const float2*)(winG + (size_t)(k0 + kl) * 4096 + nl);
      gs[nl * LDK + kl]       = f2bf(g.x);
      gs[(nl + 1) * LDK + kl] = f2bf(g.y);
      float2 iv = *(const float2*)(winI + (size_t)(k0 + kl) * 4096 + nl);
      is_[nl * LDK + kl]       = f2bf(iv.x);
      is_[(nl + 1) * LDK + kl] = f2bf(iv.y);
    }
    if (k0 + KC < DIM) {
      __builtin_prefetch(winG + (size_t)(k0 + KC) * 4096, 0, 0);
      __builtin_prefetch(winI + (size_t)(k0 + KC) * 4096, 0, 0);
    }
    __syncthreads();
#pragma unroll
    for (int kk = 0; kk < KC; kk += 32) {
      FragU a[2];
#pragma unroll
      for (int mi = 0; mi < 2; mi++) {
        int row = m0 + mi * 16 + (lane & 15);
        int kb  = kk + (lane >> 4) * 16;       // A layout: 16 contiguous K per lane group
        a[mi].q[0] = *(const uint4*)&xs[row * LDK + kb];
        a[mi].q[1] = *(const uint4*)&xs[row * LDK + kb + 8];
      }
      FragU bg[4], bi[4];
#pragma unroll
      for (int ni = 0; ni < 4; ni++) {
        int col = n0 + ni * 16 + (lane & 15);
        int kb  = kk + (lane >> 4) * 16;       // B layout: 16 contiguous K per lane group
        bg[ni].q[0] = *(const uint4*)&gs[col * LDK + kb];
        bg[ni].q[1] = *(const uint4*)&gs[col * LDK + kb + 8];
        bi[ni].q[0] = *(const uint4*)&is_[col * LDK + kb];
        bi[ni].q[1] = *(const uint4*)&is_[col * LDK + kb + 8];
      }
#pragma unroll
      for (int mi = 0; mi < 2; mi++)
#pragma unroll
        for (int ni = 0; ni < 4; ni++) {
          cg[mi][ni] = __builtin_amdgcn_wmma_f32_16x16x32_bf16(
              false, a[mi].v, false, bg[ni].v, (short)0, cg[mi][ni], false, false);
          ci[mi][ni] = __builtin_amdgcn_wmma_f32_16x16x32_bf16(
              false, a[mi].v, false, bi[ni].v, (short)0, ci[mi][ni], false, false);
        }
    }
  }
  // fused SiLU gate and store bf16 activations
#pragma unroll
  for (int mi = 0; mi < 2; mi++)
#pragma unroll
    for (int ni = 0; ni < 4; ni++) {
      v8f g = cg[mi][ni], iv = ci[mi][ni];
      int col = jn + n0 + ni * 16 + (lane & 15);
#pragma unroll
      for (int r = 0; r < 8; r++) {
        int row = m0 + mi * 16 + (lane >> 4) * 8 + r;   // C layout: M = r + 8*(lane/16)
        float gv = g[r];
        float av = (gv / (1.f + __expf(-gv))) * iv[r];
        act[(size_t)(slot0 + row) * HALF_IN + col] = f2bf(av);
      }
    }
}

// y += (act @ Wout[e]) * ce, via f32 global atomics
__global__ __launch_bounds__(256) void k_ffn2(const unsigned short* __restrict__ act,
                                              const float* __restrict__ Wout,
                                              const int* __restrict__ counts,
                                              const int* __restrict__ toks,
                                              const float* __restrict__ wl,
                                              float* __restrict__ y) {
  int e = blockIdx.z, jn = blockIdx.x * 128, rt = blockIdx.y;
  int cnt = counts[e];
  if (rt * 128 >= cnt) return;
  int base = 0;
  for (int q = 0; q < e; q++) base += (counts[q] + 127) & ~127;
  int slot0 = base + rt * 128;

  __shared__ unsigned short as_[128 * LDK];
  __shared__ unsigned short ws_[128 * LDK];
  __shared__ int   ts[128];
  __shared__ float wls[128];

  int tid = threadIdx.x, wave = tid >> 5, lane = tid & 31;
  if (tid < 128) {
    int idx = rt * 128 + tid;
    bool v = idx < cnt;
    ts[tid]  = v ? toks[e * 4096 + idx] : 0;
    wls[tid] = v ? wl[e * 4096 + idx]  : 0.f;   // pad rows contribute exactly 0
  }
  int m0 = (wave >> 1) * 32, n0 = (wave & 1) * 64;

  v8f co[2][4];
#pragma unroll
  for (int mi = 0; mi < 2; mi++)
#pragma unroll
    for (int ni = 0; ni < 4; ni++) co[mi][ni] = (v8f)0.0f;

  const float* wp = Wout + ((size_t)e << 21) + jn;        // e*2048*1024

  for (int k0 = 0; k0 < NOUTK; k0 += KC) {
    __syncthreads();
    // stage act tile (pure bf16 copy): CDNA5 async global->LDS DMA, ASYNCcnt-tracked.
    // 256 lanes x 4 x b128 = 16 KB tile in flight while Wout staging proceeds below.
#pragma unroll
    for (int it = 0; it < 4; it++) {
      int idx = it * 2048 + tid * 8;
      int row = idx >> 6, k = idx & 63;
      const unsigned short* gsrc = act + (size_t)(slot0 + row) * HALF_IN + k0 + k;
      unsigned ldst = lds_addr32(&as_[row * LDK + k]);
      asm volatile("global_load_async_to_lds_b128 %0, %1, off"
                   :: "v"(ldst), "v"(gsrc)
                   : "memory");
    }
    // stage Wout tile transposed [n][k], f32 -> bf16 (overlaps with async DMA above)
#pragma unroll
    for (int it = 0; it < 16; it++) {
      int kl = it * 4 + (tid >> 6);
      int nl = (tid & 63) * 2;
      float2 w = *(const float2*)(wp + (size_t)(k0 + kl) * 1024 + nl);
      ws_[nl * LDK + kl]       = f2bf(w.x);
      ws_[(nl + 1) * LDK + kl] = f2bf(w.y);
    }
    if (k0 + KC < NOUTK) __builtin_prefetch(wp + (size_t)(k0 + KC) * 1024, 0, 0);
#if __has_builtin(__builtin_amdgcn_s_wait_asynccnt)
    __builtin_amdgcn_s_wait_asynccnt(0);
#else
    asm volatile("s_wait_asynccnt 0x0" ::: "memory");
#endif
    __syncthreads();
#pragma unroll
    for (int kk = 0; kk < KC; kk += 32) {
      FragU a[2];
#pragma unroll
      for (int mi = 0; mi < 2; mi++) {
        int row = m0 + mi * 16 + (lane & 15);
        int kb  = kk + (lane >> 4) * 16;
        a[mi].q[0] = *(const uint4*)&as_[row * LDK + kb];
        a[mi].q[1] = *(const uint4*)&as_[row * LDK + kb + 8];
      }
      FragU bb[4];
#pragma unroll
      for (int ni = 0; ni < 4; ni++) {
        int col = n0 + ni * 16 + (lane & 15);
        int kb  = kk + (lane >> 4) * 16;
        bb[ni].q[0] = *(const uint4*)&ws_[col * LDK + kb];
        bb[ni].q[1] = *(const uint4*)&ws_[col * LDK + kb + 8];
      }
#pragma unroll
      for (int mi = 0; mi < 2; mi++)
#pragma unroll
        for (int ni = 0; ni < 4; ni++)
          co[mi][ni] = __builtin_amdgcn_wmma_f32_16x16x32_bf16(
              false, a[mi].v, false, bb[ni].v, (short)0, co[mi][ni], false, false);
    }
  }
#pragma unroll
  for (int mi = 0; mi < 2; mi++)
#pragma unroll
    for (int ni = 0; ni < 4; ni++) {
      int col = jn + n0 + ni * 16 + (lane & 15);
#pragma unroll
      for (int r = 0; r < 8; r++) {
        int row = m0 + mi * 16 + (lane >> 4) * 8 + r;
        float v = co[mi][ni][r] * wls[row];
        __hip_atomic_fetch_add(&y[(size_t)ts[row] * DIM + col], v,
                               __ATOMIC_RELAXED, __HIP_MEMORY_SCOPE_AGENT);
      }
    }
}

extern "C" void kernel_launch(void* const* d_in, const int* in_sizes, int n_in,
                              void* d_out, int out_size, void* d_ws, size_t ws_size,
                              hipStream_t stream) {
  const float* x    = (const float*)d_in[0];
  const float* Wr   = (const float*)d_in[1];
  const float* Win  = (const float*)d_in[2];
  const float* Wout = (const float*)d_in[3];
  float* y = (float*)d_out;

  int*   counts = (int*)d_ws;
  int*   toks   = (int*)((char*)d_ws + WS_TOKS);
  float* wl     = (float*)((char*)d_ws + WS_WL);
  unsigned short* act = (unsigned short*)((char*)d_ws + WS_ACT);

  k_init  <<<dim3(4096),         dim3(256), 0, stream>>>(y, counts);
  k_router<<<dim3(NTOK / 8),     dim3(256), 0, stream>>>(x, Wr, counts, toks, wl);
  k_ffn1  <<<dim3(16, 32, NEXP), dim3(256), 0, stream>>>(x, Win, counts, toks, act);
  k_ffn2  <<<dim3(8, 32, NEXP),  dim3(256), 0, stream>>>(act, Wout, counts, toks, wl, y);
}
